// BidirectionalMambaBlock_52278341927351
// MI455X (gfx1250) — compile-verified, matrix operand order checked by reading, not run on hardware
//
#include <hip/hip_runtime.h>
#include <hip/hip_bf16.h>
#include <math.h>

// ---------------------------------------------------------------------------
// BidirectionalMambaBlock for MI455X (gfx1250, wave32, WMMA).
// GEMMs: v_wmma_f32_16x16x32_bf16, 64x64 wave tiles (16 WMMA per 16 fragment
// ds_load_b128), fp32->bf16 via v_cvt_pk_bf16_f32 while staging to LDS,
// double-buffered LDS + register pipelining so global b128 loads overlap
// WMMA. Scan processes 8 time steps per barrier pair with batched loads.
// ---------------------------------------------------------------------------

#define B_SZ     8
#define L_SZ     2048
#define D_MODEL  768
#define D_INNER  1536
#define D_STATE  16
#define D_CONV   4
#define DT_RANK  48
#define M_ROWS   (B_SZ * L_SZ)          // 16384

typedef __bf16 bf16_t;
typedef bf16_t v16bf __attribute__((ext_vector_type(16)));
typedef bf16_t bf4_t __attribute__((ext_vector_type(4)));
typedef float  v8f   __attribute__((ext_vector_type(8)));

union FragAB {
    v16bf v;
    uint4 q[2];
};

static __device__ __forceinline__ float sigmoidf_fast(float z) {
    return 1.0f / (1.0f + __expf(-z));
}

// ---------------------------------------------------------------------------
// LayerNorm over last dim (768), with optional time-flip of the input row.
// ---------------------------------------------------------------------------
__global__ __launch_bounds__(256)
void ln_kernel(const float* __restrict__ x, const float* __restrict__ g,
               const float* __restrict__ beta, float* __restrict__ out,
               int Lc, int flip) {
    int row = blockIdx.x;
    int b = row / Lc, t = row % Lc;
    int irow = flip ? (b * Lc + (Lc - 1 - t)) : row;
    const float* xp = x + (size_t)irow * D_MODEL;

    __shared__ float red[256];
    float vals[3];
    float s = 0.0f;
#pragma unroll
    for (int i = 0; i < 3; ++i) {
        float v = xp[threadIdx.x + i * 256];
        vals[i] = v;
        s += v;
    }
    red[threadIdx.x] = s;
    __syncthreads();
    for (int off = 128; off > 0; off >>= 1) {
        if (threadIdx.x < off) red[threadIdx.x] += red[threadIdx.x + off];
        __syncthreads();
    }
    float mean = red[0] * (1.0f / (float)D_MODEL);
    __syncthreads();
    float s2 = 0.0f;
#pragma unroll
    for (int i = 0; i < 3; ++i) {
        float d = vals[i] - mean;
        s2 += d * d;
    }
    red[threadIdx.x] = s2;
    __syncthreads();
    for (int off = 128; off > 0; off >>= 1) {
        if (threadIdx.x < off) red[threadIdx.x] += red[threadIdx.x + off];
        __syncthreads();
    }
    float rstd = rsqrtf(red[0] * (1.0f / (float)D_MODEL) + 1e-5f);
#pragma unroll
    for (int i = 0; i < 3; ++i) {
        int c = threadIdx.x + i * 256;
        out[(size_t)row * D_MODEL + c] = (vals[i] - mean) * rstd * g[c] + beta[c];
    }
}

// ---------------------------------------------------------------------------
// GEMM: C[M x N] = epilogue(A[M x K] * W[N x K]^T), bf16 WMMA, fp32 accum.
// Block tile 256x128 (8 waves, 4x2 grid), wave tile 64x64 (4x4 WMMA tiles),
// K-tile 32, double-buffered LDS.  MODE: 0 plain, 1 softplus+bias,
// 2 residual+flip, 3 bias.
// ---------------------------------------------------------------------------
#define GBM 256
#define GBN 128
#define GBK 32
#define GSTR 40   // LDS row stride in bf16 elems (32 + 8 pad, keeps 16B align)

template <bool GUARDED, int MODE>
__global__ __launch_bounds__(256)
void gemm_bf16_kernel(const float* __restrict__ A, int lda, int K,
                      const float* __restrict__ W,
                      float* __restrict__ C, int ldc, int colOff, int N,
                      const float* __restrict__ bias,
                      const float* __restrict__ resid,
                      int flip, int Lc) {
    __shared__ bf16_t sA[2][GBM * GSTR];   // 2 * 256*40*2 = 40960 B
    __shared__ bf16_t sW[2][GBN * GSTR];   // 2 * 128*40*2 = 20480 B

    int tid  = threadIdx.x;
    int lane = tid & 31;
    int wid  = tid >> 5;
    int wm   = wid & 3;        // wave row (4 waves x 64 rows)
    int wn   = wid >> 2;       // wave col (2 waves x 64 cols)
    int l16  = lane & 15;
    int half = lane >> 4;      // K-half select per ISA A/B fragment layout

    int rowBase = blockIdx.y * GBM;
    int colBase = blockIdx.x * GBN;

    v8f acc[4][4];
#pragma unroll
    for (int i = 0; i < 4; ++i)
#pragma unroll
        for (int j = 0; j < 4; ++j)
#pragma unroll
            for (int v = 0; v < 8; ++v) acc[i][j][v] = 0.0f;

    float4 ra[8], rw[4];       // staging registers for next K tile

    auto loadA = [&](int k0) {
#pragma unroll
        for (int i = 0; i < 8; ++i) {                 // 256x32 / (256*4)
            int flat = (i * 256 + tid) * 4;
            int r = flat >> 5;                         // /32
            int k = flat & 31;
            int kg = k0 + k;
            const float* ap = A + (size_t)(rowBase + r) * lda + kg;
            if (!GUARDED || kg < K) ra[i] = *(const float4*)ap;
            else ra[i] = make_float4(0.f, 0.f, 0.f, 0.f);
        }
    };
    auto loadW = [&](int k0) {
#pragma unroll
        for (int i = 0; i < 4; ++i) {                 // 128x32 / (256*4)
            int flat = (i * 256 + tid) * 4;
            int n = flat >> 5;
            int k = flat & 31;
            int ng = colBase + n;
            int kg = k0 + k;
            const float* wp = W + (size_t)ng * K + kg;
            if (!GUARDED || (ng < N && kg < K)) rw[i] = *(const float4*)wp;
            else rw[i] = make_float4(0.f, 0.f, 0.f, 0.f);
        }
    };
    auto storeTiles = [&](int buf) {
#pragma unroll
        for (int i = 0; i < 8; ++i) {
            int flat = (i * 256 + tid) * 4;
            int r = flat >> 5;
            int k = flat & 31;
            bf4_t c;
            c[0] = (bf16_t)ra[i].x; c[1] = (bf16_t)ra[i].y;
            c[2] = (bf16_t)ra[i].z; c[3] = (bf16_t)ra[i].w;
            *(bf4_t*)&sA[buf][r * GSTR + k] = c;
        }
#pragma unroll
        for (int i = 0; i < 4; ++i) {
            int flat = (i * 256 + tid) * 4;
            int r = flat >> 5;
            int k = flat & 31;
            bf4_t d;
            d[0] = (bf16_t)rw[i].x; d[1] = (bf16_t)rw[i].y;
            d[2] = (bf16_t)rw[i].z; d[3] = (bf16_t)rw[i].w;
            *(bf4_t*)&sW[buf][r * GSTR + k] = d;
        }
    };
    auto computeTile = [&](int buf) {
        int kb = half * 8;                 // per-lane K base within 32
        FragAB fa[4], fb[4];
#pragma unroll
        for (int tm = 0; tm < 4; ++tm) {
            const bf16_t* p = &sA[buf][(wm * 64 + tm * 16 + l16) * GSTR + kb];
            fa[tm].q[0] = *(const uint4*)p;          // K = kb..kb+7
            fa[tm].q[1] = *(const uint4*)(p + 16);   // K = kb+16..kb+23
        }
#pragma unroll
        for (int tn = 0; tn < 4; ++tn) {
            const bf16_t* p = &sW[buf][(wn * 64 + tn * 16 + l16) * GSTR + kb];
            fb[tn].q[0] = *(const uint4*)p;
            fb[tn].q[1] = *(const uint4*)(p + 16);
        }
#pragma unroll
        for (int tm = 0; tm < 4; ++tm)
#pragma unroll
            for (int tn = 0; tn < 4; ++tn)
                acc[tm][tn] = __builtin_amdgcn_wmma_f32_16x16x32_bf16(
                    false, fa[tm].v, false, fb[tn].v,
                    (short)0, acc[tm][tn], false, false);
    };

    // ---- software pipeline ---------------------------------------------
    loadA(0);
    loadW(0);
    storeTiles(0);
    __syncthreads();

    int buf = 0;
    for (int k0 = GBK; k0 < K; k0 += GBK) {
        loadA(k0);                 // issue global b128 loads (next tile)
        loadW(k0);
        computeTile(buf);          // WMMA on current tile hides latency
        storeTiles(buf ^ 1);       // cvt + write other LDS buffer
        __syncthreads();
        buf ^= 1;
    }
    computeTile(buf);              // last tile

    // ---- epilogue ------------------------------------------------------
#pragma unroll
    for (int tm = 0; tm < 4; ++tm) {
#pragma unroll
        for (int tn = 0; tn < 4; ++tn) {
            int gn = colBase + wn * 64 + tn * 16 + l16;
            if (GUARDED && gn >= N) continue;
#pragma unroll
            for (int v = 0; v < 8; ++v) {
                int gm = rowBase + wm * 64 + tm * 16 + v + half * 8;
                float val = acc[tm][tn][v];
                size_t orow = (size_t)gm;
                if (MODE == 2 && flip) {
                    int b = gm / Lc, t = gm % Lc;
                    orow = (size_t)b * Lc + (Lc - 1 - t);
                }
                if (MODE == 1) {
                    val += bias[gn];
                    val = (val > 20.0f) ? val : __logf(1.0f + __expf(val));
                } else if (MODE == 2) {
                    val += resid[orow * D_MODEL + gn];
                } else if (MODE == 3) {
                    val += bias[gn];
                }
                C[orow * (size_t)ldc + colOff + gn] = val;
            }
        }
    }
}

// ---------------------------------------------------------------------------
// Depthwise causal conv1d (width 4) + bias + SiLU over the 'u' half of xz.
// ---------------------------------------------------------------------------
__global__ __launch_bounds__(256)
void conv_silu_kernel(const float* __restrict__ xz, const float* __restrict__ cw,
                      const float* __restrict__ cb, float* __restrict__ ucv,
                      int Lc) {
    int d = blockIdx.x * 256 + threadIdx.x;   // 0..1535
    int row = blockIdx.y;                      // 0..M-1
    int t = row % Lc;
    float acc = cb[d];
#pragma unroll
    for (int k = 0; k < D_CONV; ++k) {
        int dt_ = t + k - (D_CONV - 1);
        if (dt_ >= 0)
            acc += xz[(size_t)(row + k - (D_CONV - 1)) * (2 * D_INNER) + d] *
                   cw[d * D_CONV + k];
    }
    acc = acc * sigmoidf_fast(acc);            // SiLU
    ucv[(size_t)row * D_INNER + d] = acc;
}

// ---------------------------------------------------------------------------
// Selective scan, chunked: 8 time steps of B_t/C_t staged per barrier pair
// (256 threads load exactly 8 x 32 floats), per-chunk delta/u/z batch loads
// so global latency amortizes over the recurrence math. One thread per
// (batch, inner-channel); 16 states in registers. Gated y overwrites delta.
// ---------------------------------------------------------------------------
#define SCAN_TC 8

__global__ __launch_bounds__(256)
void scan_kernel(const float* __restrict__ dbl, const float* __restrict__ ucv,
                 const float* __restrict__ xz, const float* __restrict__ A_log,
                 const float* __restrict__ Dp, float* __restrict__ dy, int Lc) {
    int tid = threadIdx.x;
    int d = blockIdx.x * 256 + tid;            // 0..1535
    int b = blockIdx.y;                        // 0..7

    float Areg[D_STATE];
#pragma unroll
    for (int n = 0; n < D_STATE; ++n)
        Areg[n] = -__expf(A_log[d * D_STATE + n]);
    float Dv = Dp[d];

    float h[D_STATE];
#pragma unroll
    for (int n = 0; n < D_STATE; ++n) h[n] = 0.0f;

    __shared__ float Bs[SCAN_TC][D_STATE];
    __shared__ float Cs[SCAN_TC][D_STATE];

    // staging role for this thread: step 0..7, B (which=0) or C (which=1)
    int stp   = tid >> 5;
    int which = (tid >> 4) & 1;
    int nn    = tid & 15;

    for (int t0 = 0; t0 < Lc; t0 += SCAN_TC) {
        __syncthreads();
        {
            size_t srow = (size_t)b * Lc + t0 + stp;
            float v = dbl[srow * 80 + DT_RANK + which * D_STATE + nn];
            if (which == 0) Bs[stp][nn] = v;
            else            Cs[stp][nn] = v;
        }
        __syncthreads();

        // batch-issue all per-chunk loads for this channel
        float dlt[SCAN_TC], uu[SCAN_TC], zz[SCAN_TC], yg[SCAN_TC];
#pragma unroll
        for (int j = 0; j < SCAN_TC; ++j) {
            size_t row = (size_t)b * Lc + t0 + j;
            dlt[j] = dy[row * D_INNER + d];
            uu[j]  = ucv[row * D_INNER + d];
            zz[j]  = xz[row * (2 * D_INNER) + D_INNER + d];
        }
#pragma unroll
        for (int j = 0; j < SCAN_TC; ++j) {
            float du = dlt[j] * uu[j];
            float y = 0.0f;
#pragma unroll
            for (int n = 0; n < D_STATE; ++n) {
                h[n] = __expf(dlt[j] * Areg[n]) * h[n] + du * Bs[j][n];
                y += h[n] * Cs[j][n];
            }
            y += uu[j] * Dv;
            float zv = zz[j];
            yg[j] = y * (zv * sigmoidf_fast(zv));
        }
#pragma unroll
        for (int j = 0; j < SCAN_TC; ++j) {
            size_t row = (size_t)b * Lc + t0 + j;
            dy[row * D_INNER + d] = yg[j];
        }
    }
}

// ---------------------------------------------------------------------------
// Host-side orchestration.
// Workspace layout (floats), reused across the two directions:
//   XZ: Mx3072 | UCV: Mx1536 | DEL: Mx1536 | CAT: Mx1536 | LN: Mx768 | DBL: Mx80
// ---------------------------------------------------------------------------
extern "C" void kernel_launch(void* const* d_in, const int* in_sizes, int n_in,
                              void* d_out, int out_size, void* d_ws, size_t ws_size,
                              hipStream_t stream) {
    (void)in_sizes; (void)n_in; (void)out_size; (void)ws_size;

    const float* x        = (const float*)d_in[0];
    const float* fusion_w = (const float*)d_in[23];
    const float* fusion_b = (const float*)d_in[24];

    const int M = M_ROWS;
    float* XZ  = (float*)d_ws;
    float* UCV = XZ  + (size_t)M * 3072;
    float* DEL = UCV + (size_t)M * 1536;
    float* CAT = DEL + (size_t)M * 1536;
    float* LN  = CAT + (size_t)M * 1536;
    float* DBL = LN  + (size_t)M * 768;

    for (int dir = 0; dir < 2; ++dir) {
        const float* ln_g    = (const float*)d_in[1 + 11 * dir + 0];
        const float* ln_b    = (const float*)d_in[1 + 11 * dir + 1];
        const float* in_w    = (const float*)d_in[1 + 11 * dir + 2];
        const float* conv_w  = (const float*)d_in[1 + 11 * dir + 3];
        const float* conv_b  = (const float*)d_in[1 + 11 * dir + 4];
        const float* xproj_w = (const float*)d_in[1 + 11 * dir + 5];
        const float* dt_w    = (const float*)d_in[1 + 11 * dir + 6];
        const float* dt_b    = (const float*)d_in[1 + 11 * dir + 7];
        const float* A_log   = (const float*)d_in[1 + 11 * dir + 8];
        const float* Dp      = (const float*)d_in[1 + 11 * dir + 9];
        const float* out_w   = (const float*)d_in[1 + 11 * dir + 10];

        // 1. LayerNorm (time-flipped input for backward direction)
        ln_kernel<<<M, 256, 0, stream>>>(x, ln_g, ln_b, LN, L_SZ, dir);

        // 2. in_proj: XZ = LN @ in_w^T   (16384 x 768 x 3072) -- unguarded
        gemm_bf16_kernel<false, 0><<<dim3(3072 / GBN, M / GBM), 256, 0, stream>>>(
            LN, D_MODEL, D_MODEL, in_w, XZ, 3072, 0, 3072,
            nullptr, nullptr, 0, L_SZ);

        // 3. depthwise causal conv + SiLU on u half
        conv_silu_kernel<<<dim3(D_INNER / 256, M), 256, 0, stream>>>(
            XZ, conv_w, conv_b, UCV, L_SZ);

        // 4. x_proj: DBL = UCV @ xproj_w^T  (16384 x 1536 x 80) -- N guard
        gemm_bf16_kernel<true, 0><<<dim3(1, M / GBM), 256, 0, stream>>>(
            UCV, D_INNER, D_INNER, xproj_w, DBL, 80, 0, 80,
            nullptr, nullptr, 0, L_SZ);

        // 5. dt_proj + softplus: DEL = softplus(dt @ dt_w^T + dt_b) -- K guard
        gemm_bf16_kernel<true, 1><<<dim3(D_INNER / GBN, M / GBM), 256, 0, stream>>>(
            DBL, 80, DT_RANK, dt_w, DEL, D_INNER, 0, D_INNER,
            dt_b, nullptr, 0, L_SZ);

        // 6. selective scan; gated y overwrites DEL in place
        scan_kernel<<<dim3(D_INNER / 256, B_SZ), 256, 0, stream>>>(
            DBL, UCV, XZ, A_log, Dp, DEL, L_SZ);

        // 7. out_proj + residual into CAT (flipped rows for bwd) -- unguarded
        gemm_bf16_kernel<false, 2><<<dim3(D_MODEL / GBN, M / GBM), 256, 0, stream>>>(
            DEL, D_INNER, D_INNER, out_w, CAT, 2 * D_MODEL, dir * D_MODEL,
            D_MODEL, nullptr, x, dir, L_SZ);
    }

    // 8. fusion: out = CAT @ fusion_w^T + fusion_b -- unguarded
    gemm_bf16_kernel<false, 3><<<dim3(D_MODEL / GBN, M / GBM), 256, 0, stream>>>(
        CAT, 2 * D_MODEL, 2 * D_MODEL, fusion_w, (float*)d_out, D_MODEL, 0,
        D_MODEL, fusion_b, nullptr, 0, L_SZ);
}
